// GearboxCNNLSTM_28458453303296
// MI455X (gfx1250) — matrix operand order repaired and checked by reading, not compiled, not run on hardware
//
#include <hip/hip_runtime.h>
#include <hip/hip_bf16.h>

typedef __attribute__((ext_vector_type(16))) _Float16 v16h;
typedef __attribute__((ext_vector_type(8)))  float    v8f;

// Native CDNA5 transcendental (V_TANH_F32, TRANS32 class). v_nop covers the
// trans-op result-use hazard since the compiler cannot see into the asm.
__device__ __forceinline__ float fast_tanh(float x) {
  float r;
  asm("v_tanh_f32 %0, %1\n\tv_nop" : "=v"(r) : "v"(x));
  return r;
}
__device__ __forceinline__ float sigf(float x) { return 0.5f * fast_tanh(0.5f * x) + 0.5f; }

// ---- WMMA fragment helpers ----
// All WMMA operands live in LDS in fragment-major layout: frag[lane*16 + e],
// 32B per lane -> each operand load is 2x ds_load_b128.
// A-fragment map (16Mx32K f16): m=lane&15, h=lane>>4, k=(e<8?e:e+8)+h*8
// B-fragment map (32Kx16N f16): n=lane&15, h=lane>>4, k=(e<8?e:e+8)+h*8
__device__ __forceinline__ v16h ldfrag(const _Float16* frag, int lane) {
  return *(const v16h*)(frag + lane * 16);
}
// inverse of the A-map k -> (h,e), for scattering results into A fragments
__device__ __forceinline__ void a_inv(int k, int& h, int& e) {
  if (k < 8)       { h = 0; e = k; }
  else if (k < 16) { h = 1; e = k - 8; }
  else if (k < 24) { h = 0; e = k - 8; }
  else             { h = 1; e = k - 16; }
}
// C/D 16x16 f32: lane 0-15 -> N=lane, rows M=r (+8 for upper lane half)
__device__ __forceinline__ void stC_f32(float* base, int ldn, int lane, v8f c) {
  int n = lane & 15, mh = (lane >> 4) * 8;
#pragma unroll
  for (int r = 0; r < 8; ++r) base[(mh + r) * ldn + n] = c[r];
}
__device__ __forceinline__ v8f wmma_f16(v16h a, v16h b, v8f c) {
  return __builtin_amdgcn_wmma_f32_16x16x32_f16(false, a, false, b, (short)0, c, false, false);
}

// =====================================================================
// K1: fused depthwise conv(k=5,pad2) + BN + ReLU + maxpool(256) -> y(B,72,16)
// grid (16, B), block 288 (= 72 channels * 4 position-quarters)
// =====================================================================
__global__ __launch_bounds__(288) void conv_pool_kernel(
    const float* __restrict__ x, const float* __restrict__ cw,
    const float* __restrict__ cb, const float* __restrict__ gamma,
    const float* __restrict__ beta, float* __restrict__ y) {
  __shared__ float xs[9 * 264];
  __shared__ float red[72 * 4];
  const int tid = threadIdx.x;
  const int t = blockIdx.x, b = blockIdx.y;
  const int wbase = t * 256 - 2;
  for (int q = tid; q < 260 * 9; q += 288) {
    int wo = q / 9, ch = q % 9;
    int w = wbase + wo;
    float v = 0.f;
    if (w >= 0 && w < 4096) v = x[((size_t)b * 4096 + w) * 9 + ch];
    xs[ch * 264 + wo] = v;
  }
  if (t < 15 && tid < 32) {  // prefetch next tile into GL2 (global_prefetch)
    const float* nx = &x[((size_t)b * 4096 + (t + 1) * 256) * 9];
    __builtin_prefetch(nx + tid * 64, 0, 1);
  }
  __syncthreads();
  const int c = tid >> 2, part = tid & 3;
  const int ic = c >> 3;
  const float w0 = cw[c * 5 + 0], w1 = cw[c * 5 + 1], w2 = cw[c * 5 + 2],
              w3 = cw[c * 5 + 3], w4 = cw[c * 5 + 4];
  const float scale = gamma[c] * rsqrtf(1.0f + 1e-5f);
  const float bet = beta[c], bias = cb[c];
  const float* xr = xs + ic * 264 + part * 64;
  float mx = -1e30f;
#pragma unroll 4
  for (int p = 0; p < 64; ++p) {
    float s = w0 * xr[p] + w1 * xr[p + 1] + w2 * xr[p + 2] + w3 * xr[p + 3] + w4 * xr[p + 4];
    float v = (s + bias) * scale + bet;
    mx = fmaxf(mx, v);
  }
  red[c * 4 + part] = mx;
  __syncthreads();
  if (tid < 72) {
    float m = fmaxf(fmaxf(red[tid * 4 + 0], red[tid * 4 + 1]),
                    fmaxf(red[tid * 4 + 2], red[tid * 4 + 3]));
    y[((size_t)b * 72 + tid) * 16 + t] = fmaxf(m, 0.f);
  }
}

// =====================================================================
// K2: sensor attention; emit seq as pre-swizzled A fragments
//     seq_g layout: [bblk(32)][t(16)][kt(3)][512]  (b = bblk*16 + m)
//     raw plain f16 (B,1024). grid B, block 256.
// =====================================================================
__global__ __launch_bounds__(256) void attn_combine_kernel(
    const float* __restrict__ y, const float* __restrict__ aw1,
    const float* __restrict__ ab1, const float* __restrict__ aw2,
    const float* __restrict__ ab2, float* __restrict__ out_attn,
    _Float16* __restrict__ seq_g, _Float16* __restrict__ raw_out) {
  __shared__ float yb[72 * 16];
  __shared__ float a1[8 * 32];
  __shared__ float lg[8];
  __shared__ float at[8];
  const int tid = threadIdx.x, b = blockIdx.x;
  const int bblk = b >> 4, m = b & 15;
  for (int i = tid; i < 1152; i += 256) yb[i] = y[(size_t)b * 1152 + i];
  __syncthreads();
  {  // a1[s][o] = tanh(raw[s] . aw1[o] + ab1[o]); one (s,o) per thread
    int s = tid >> 5, o = tid & 31;
    float z = ab1[o];
    const float* w = aw1 + (o << 7);
    for (int f = 0; f < 128; ++f) {
      int j = f >> 4, tt = f & 15;
      z += yb[(s * 8 + j) * 16 + tt] * w[f];
    }
    a1[tid] = fast_tanh(z);
  }
  __syncthreads();
  if (tid < 8) {
    float z = ab2[0];
    for (int o = 0; o < 32; ++o) z += a1[tid * 32 + o] * aw2[o];
    lg[tid] = z;
  }
  __syncthreads();
  if (tid < 8) {  // softmax over 8 sensors
    float mx = -1e30f;
    for (int s = 0; s < 8; ++s) mx = fmaxf(mx, lg[s]);
    float sum = 0.f;
    for (int s = 0; s < 8; ++s) sum += __expf(lg[s] - mx);
    float v = __expf(lg[tid] - mx) / sum;
    at[tid] = v;
    out_attn[(size_t)b * 8 + tid] = v;
  }
  __syncthreads();
  // seq as A fragments: i = ((t*3+kt)*2 + h)*16 + e
  for (int i = tid; i < 1536; i += 256) {
    int e = i & 15, h = (i >> 4) & 1, fp = i >> 5;  // fp = t*3+kt
    int t = fp / 3, kt = fp % 3;
    int kk = (e < 8 ? e : e + 8) + h * 8;
    int k = kt * 32 + kk;  // padded channel 0..95
    float v = 0.f;
    if (k < 72) {
      v = yb[k * 16 + t];
      if (k < 64) v *= at[k >> 3];
    }
    int lane = m + h * 16;
    seq_g[((size_t)(bblk * 16 + t) * 3 + kt) * 512 + lane * 16 + e] = (_Float16)v;
  }
  // raw plain: raw[s][j*16+t] = y[s*8+j][t]
  for (int i = tid; i < 1024; i += 256) {
    int s = i >> 7, f = i & 127, j = f >> 4, tt = f & 15;
    raw_out[(size_t)b * 1024 + i] = (_Float16)yb[(s * 8 + j) * 16 + tt];
  }
}

// =====================================================================
// K3: bidirectional LSTM (WMMA gate GEMMs) + temporal attention + context
//     + fault head. grid 32 (16 batch rows each), block 256 (8 waves).
// =====================================================================
__global__ __launch_bounds__(256) void lstm_kernel(
    const _Float16* __restrict__ seq_g, const float* __restrict__ wihf,
    const float* __restrict__ whhf, const float* __restrict__ bf,
    const float* __restrict__ wihr, const float* __restrict__ whhr,
    const float* __restrict__ br, const float* __restrict__ tw1,
    const float* __restrict__ tb1, const float* __restrict__ tw2,
    const float* __restrict__ tb2, const float* __restrict__ fw1,
    const float* __restrict__ fb1, const float* __restrict__ fw2,
    const float* __restrict__ fb2, float* __restrict__ ctx_out,
    float* __restrict__ out_fault, float* __restrict__ out_tw) {
  __shared__ __align__(32) _Float16 seqf[16 * 3 * 512];  // 48 KB  A frags [t][kt]
  __shared__ __align__(32) _Float16 wihf_s[8 * 3 * 512]; // 24 KB  B frags [nt][kt]
  __shared__ __align__(32) _Float16 whhf_s[8 * 512];     //  8 KB  B frags [nt]
  __shared__ __align__(32) _Float16 hAf[512];            //  1 KB  h as A frag
  __shared__ float    gates[16 * 128];                   //  8 KB
  __shared__ float    bbs[128];
  __shared__ float    cst[16 * 32];                      //  2 KB cell state
  __shared__ _Float16 hs[2 * 16 * 16 * 32];              // 32 KB [dir][t][m][j]
  __shared__ float    sc[16 * 16];
  __shared__ float    ctx[16 * 64];
  __shared__ float    fdh[16 * 32];
  __shared__ float    tw1s[16 * 64];                     //  4 KB
  __shared__ float    fw1s[32 * 64];                     //  8 KB
  const int tid = threadIdx.x;
  const int b0 = blockIdx.x * 16;
  for (int idx = tid; idx < 16 * 3 * 512; idx += 256)
    seqf[idx] = seq_g[(size_t)blockIdx.x * (16 * 3 * 512) + idx];
  for (int idx = tid; idx < 16 * 64; idx += 256) tw1s[idx] = tw1[idx];
  for (int idx = tid; idx < 32 * 64; idx += 256) fw1s[idx] = fw1[idx];
  const int wv = tid >> 5, lane = tid & 31, n0 = wv * 16;
  for (int d = 0; d < 2; ++d) {
    const float* wih = d ? wihr : wihf;
    const float* whh = d ? whhr : whhf;
    const float* bb = d ? br : bf;
    __syncthreads();
    // B fragments for w_ih^T (96x128, K padded) and w_hh^T (32x128)
    for (int idx = tid; idx < 8 * 3 * 512; idx += 256) {
      int e = idx & 15, ln = (idx >> 4) & 31, fp = idx >> 9;  // fp = nt*3+kt
      int nt = fp / 3, kt = fp % 3;
      int n = nt * 16 + (ln & 15);
      int k = kt * 32 + (ln >> 4) * 8 + (e < 8 ? e : e + 8);
      wihf_s[idx] = (k < 72) ? (_Float16)wih[n * 72 + k] : (_Float16)0.f;
    }
    for (int idx = tid; idx < 8 * 512; idx += 256) {
      int e = idx & 15, ln = (idx >> 4) & 31, nt = idx >> 9;
      int n = nt * 16 + (ln & 15);
      int k = (ln >> 4) * 8 + (e < 8 ? e : e + 8);
      whhf_s[idx] = (_Float16)whh[n * 32 + k];
    }
    if (tid < 128) bbs[tid] = bb[tid];
    for (int idx = tid; idx < 512; idx += 256) hAf[idx] = (_Float16)0.f;
    for (int idx = tid; idx < 16 * 32; idx += 256) cst[idx] = 0.f;
    __syncthreads();
    for (int step = 0; step < 16; ++step) {
      const int t = d ? (15 - step) : step;
      const _Float16* At = seqf + t * 3 * 512;
      v8f acc = {};
      acc = wmma_f16(ldfrag(At + 0 * 512, lane), ldfrag(wihf_s + (wv * 3 + 0) * 512, lane), acc);
      acc = wmma_f16(ldfrag(At + 1 * 512, lane), ldfrag(wihf_s + (wv * 3 + 1) * 512, lane), acc);
      acc = wmma_f16(ldfrag(At + 2 * 512, lane), ldfrag(wihf_s + (wv * 3 + 2) * 512, lane), acc);
      acc = wmma_f16(ldfrag(hAf, lane),          ldfrag(whhf_s + wv * 512, lane), acc);
      stC_f32(gates + n0, 128, lane, acc);
      __syncthreads();
      for (int idx = tid; idx < 512; idx += 256) {
        int m = idx >> 5, j = idx & 31;
        float gi = gates[m * 128 + j] + bbs[j];
        float gf = gates[m * 128 + 32 + j] + bbs[32 + j];
        float gg = gates[m * 128 + 64 + j] + bbs[64 + j];
        float go = gates[m * 128 + 96 + j] + bbs[96 + j];
        float cn = sigf(gf) * cst[m * 32 + j] + sigf(gi) * fast_tanh(gg);
        float h = sigf(go) * fast_tanh(cn);
        cst[m * 32 + j] = cn;
        int hh, ee;
        a_inv(j, hh, ee);
        hAf[(m + hh * 16) * 16 + ee] = (_Float16)h;
        hs[((d * 16 + t) * 16 + m) * 32 + j] = (_Float16)h;
      }
      __syncthreads();
    }
  }
  {  // temporal attention scores: one (m,t) per thread
    int m = tid >> 4, t = tid & 15;
    float s = tb2[0];
    for (int o = 0; o < 16; ++o) {
      float z = tb1[o];
      for (int k = 0; k < 32; ++k) {
        z += (float)hs[((0 * 16 + t) * 16 + m) * 32 + k] * tw1s[o * 64 + k];
        z += (float)hs[((1 * 16 + t) * 16 + m) * 32 + k] * tw1s[o * 64 + 32 + k];
      }
      s += fast_tanh(z) * tw2[o];
    }
    sc[m * 16 + t] = s;
  }
  __syncthreads();
  if (tid < 16) {  // softmax over 16 timesteps for row m=tid
    int m = tid;
    float mx = -1e30f;
    for (int t = 0; t < 16; ++t) mx = fmaxf(mx, sc[m * 16 + t]);
    float sum = 0.f;
    for (int t = 0; t < 16; ++t) {
      float e = __expf(sc[m * 16 + t] - mx);
      sc[m * 16 + t] = e;
      sum += e;
    }
    float inv = 1.0f / sum;
    for (int t = 0; t < 16; ++t) {
      float w = sc[m * 16 + t] * inv;
      sc[m * 16 + t] = w;
      out_tw[(size_t)(b0 + m) * 16 + t] = w;
    }
  }
  __syncthreads();
  for (int idx = tid; idx < 16 * 64; idx += 256) {  // context
    int m = idx >> 6, k = idx & 63;
    int dsel = k >> 5, kk = k & 31;
    float acc = 0.f;
    for (int t = 0; t < 16; ++t)
      acc += sc[m * 16 + t] * (float)hs[((dsel * 16 + t) * 16 + m) * 32 + kk];
    ctx[idx] = acc;
    ctx_out[(size_t)(b0 + m) * 64 + k] = acc;
  }
  __syncthreads();
  {  // fault hidden layer: (m, 2 units) per thread
    int m = tid >> 4, u0 = (tid & 15) * 2;
    for (int uu = 0; uu < 2; ++uu) {
      int u = u0 + uu;
      float z = fb1[u];
      for (int k = 0; k < 64; ++k) z += ctx[m * 64 + k] * fw1s[u * 64 + k];
      fdh[m * 32 + u] = fmaxf(z, 0.f);
    }
  }
  __syncthreads();
  if (tid < 16) {
    int m = tid;
    float z = fb2[0];
    for (int u = 0; u < 32; ++u) z += fdh[m * 32 + u] * fw2[u];
    out_fault[b0 + m] = sigf(z);
  }
}

// =====================================================================
// K4: cross MLP + per-sensor sfe + heads, all on WMMA (wave = sensor).
// grid 32, block 256 (8 waves). 32KB phase-scratch reused across stages.
// =====================================================================
__global__ __launch_bounds__(256) void cross_heads_kernel(
    const _Float16* __restrict__ raw, const float* __restrict__ cw1,
    const float* __restrict__ cb1, const float* __restrict__ cw2,
    const float* __restrict__ cb2, const float* __restrict__ sw,
    const float* __restrict__ sb, const float* __restrict__ hw1,
    const float* __restrict__ hb1, const float* __restrict__ hw2,
    const float* __restrict__ hb2, const float* __restrict__ hw3,
    const float* __restrict__ hb3, const float* __restrict__ ctx_in,
    float* __restrict__ out_anom) {
  __shared__ __align__(32) _Float16 rawAf[32 * 512];   // 32 KB A frags of raw [kt]
  __shared__ __align__(32) _Float16 cfA[8 * 5 * 512];  // 40 KB A frags of cf [s][kt]
  __shared__ __align__(32) _Float16 gAf[4 * 512];      //  4 KB A frags of g [kt]
  __shared__ __align__(32) _Float16 h1A[8 * 2 * 512];  // 16 KB A frags of h1 [s][kt]
  __shared__ float ctx[16 * 64];                       //  4 KB
  __shared__ float gct[16 * 64];                       //  4 KB
  __shared__ float h2p[16 * 8 * 32];                   // 16 KB
  __shared__ __align__(32) char scratch[32 * 1024];    // 32 KB phase scratch
  const int tid = threadIdx.x;
  const int b0 = blockIdx.x * 16;
  const int wv = tid >> 5, lane = tid & 31, n0 = wv * 16;
  const int mh = (lane >> 4) * 8;
  // ---- phase 0: load raw (plain) + ctx; build raw A fragments ----
  _Float16* rawp = (_Float16*)scratch;  // 16*1024
  for (int idx = tid; idx < 16 * 1024; idx += 256)
    rawp[idx] = raw[(size_t)(b0 + (idx >> 10)) * 1024 + (idx & 1023)];
  for (int idx = tid; idx < 16 * 64; idx += 256)
    ctx[idx] = ctx_in[(size_t)(b0 + (idx >> 6)) * 64 + (idx & 63)];
  __syncthreads();
  for (int idx = tid; idx < 32 * 512; idx += 256) {
    int e = idx & 15, ln = (idx >> 4) & 31, kt = idx >> 9;
    int m = ln & 15, h = ln >> 4;
    int k = kt * 32 + (e < 8 ? e : e + 8) + h * 8;
    rawAf[idx] = rawp[m * 1024 + k];
  }
  __syncthreads();
  // ---- phase 1: cross layer 1 (16x1024 @ 1024->128), K chunked ----
  _Float16* bchf = (_Float16*)scratch;           // 8*512 per chunk
  _Float16* w2f = (_Float16*)(scratch + 8192);   // 16*512 B frags of cw2 [nt][kt]
  for (int idx = tid; idx < 4 * 4 * 512; idx += 256) {
    int e = idx & 15, ln = (idx >> 4) & 31, fp = idx >> 9;
    int nt = fp >> 2, kt = fp & 3;
    int n = nt * 16 + (ln & 15);
    int k = kt * 32 + (ln >> 4) * 8 + (e < 8 ? e : e + 8);
    w2f[idx] = (_Float16)cw2[n * 128 + k];
  }
  v8f acc = {};
  for (int kc = 0; kc < 32; ++kc) {
    __syncthreads();
    for (int idx = tid; idx < 8 * 512; idx += 256) {
      int e = idx & 15, ln = (idx >> 4) & 31, nt = idx >> 9;
      int n = nt * 16 + (ln & 15);
      int k = kc * 32 + (ln >> 4) * 8 + (e < 8 ? e : e + 8);
      bchf[idx] = (_Float16)cw1[(size_t)n * 1024 + k];
    }
    __syncthreads();
    acc = wmma_f16(ldfrag(rawAf + kc * 512, lane), ldfrag(bchf + wv * 512, lane), acc);
  }
  {  // bias + relu -> g A fragments
    int n = n0 + (lane & 15);
    int kt = n >> 5, hh, ee;
    a_inv(n & 31, hh, ee);
#pragma unroll
    for (int r = 0; r < 8; ++r)
      gAf[kt * 512 + (mh + r + hh * 16) * 16 + ee] = (_Float16)fmaxf(acc[r] + cb1[n], 0.f);
  }
  __syncthreads();
  // ---- cross layer 2 (128->64) on waves 0..3 ----
  if (wv < 4) {
    v8f a2 = {};
#pragma unroll
    for (int kk = 0; kk < 4; ++kk)
      a2 = wmma_f16(ldfrag(gAf + kk * 512, lane), ldfrag(w2f + (wv * 4 + kk) * 512, lane), a2);
    int n = (lane & 15) + n0;
#pragma unroll
    for (int r = 0; r < 8; ++r) gct[(mh + r) * 64 + n] = a2[r] + cb2[n];
  }
  __syncthreads();
  // ---- phase 2: build cfA (kt0/1 from ctx, kt3/4 from gct) + sfe WMMA ----
  for (int idx = tid; idx < 8 * 2 * 512; idx += 256) {
    int e = idx & 15, ln = (idx >> 4) & 31, fp = idx >> 9;
    int s = fp >> 1, kt = fp & 1;
    int m = ln & 15, h = ln >> 4;
    int k = kt * 32 + (e < 8 ? e : e + 8) + h * 8;  // 0..63
    cfA[(s * 5 + kt) * 512 + (ln * 16 + e)] = (_Float16)ctx[m * 64 + k];
  }
  for (int idx = tid; idx < 8 * 2 * 512; idx += 256) {
    int e = idx & 15, ln = (idx >> 4) & 31, fp = idx >> 9;
    int s = fp >> 1, kt = fp & 1;
    int m = ln & 15, h = ln >> 4;
    int k = kt * 32 + (e < 8 ? e : e + 8) + h * 8;  // 0..63 within g part
    cfA[(s * 5 + 3 + kt) * 512 + (ln * 16 + e)] = (_Float16)gct[m * 64 + k];
  }
  _Float16* sfb = (_Float16*)scratch;  // 8s*2nt*512 per K chunk
  v8f s0 = {}, s1 = {};
  for (int kt = 0; kt < 4; ++kt) {
    __syncthreads();
    for (int idx = tid; idx < 8 * 2 * 512; idx += 256) {
      int e = idx & 15, ln = (idx >> 4) & 31, fp = idx >> 9;
      int s = fp >> 1, nt = fp & 1;
      int n = nt * 16 + (ln & 15);
      int k = kt * 32 + (ln >> 4) * 8 + (e < 8 ? e : e + 8);
      sfb[idx] = (_Float16)sw[(s * 32 + n) * 128 + k];
    }
    __syncthreads();
    v16h av = ldfrag(rawAf + (wv * 4 + kt) * 512, lane);
    s0 = wmma_f16(av, ldfrag(sfb + (wv * 2 + 0) * 512, lane), s0);
    s1 = wmma_f16(av, ldfrag(sfb + (wv * 2 + 1) * 512, lane), s1);
  }
  {  // sfe relu+bias -> cfA kt=2 A fragments (k = 64..95 of cf)
#pragma unroll
    for (int nt = 0; nt < 2; ++nt) {
      v8f& a = nt ? s1 : s0;
      int j = nt * 16 + (lane & 15);  // 0..31
      int hh, ee;
      a_inv(j, hh, ee);
#pragma unroll
      for (int r = 0; r < 8; ++r)
        cfA[(wv * 5 + 2) * 512 + (mh + r + hh * 16) * 16 + ee] =
            (_Float16)fmaxf(a[r] + sb[wv * 32 + j], 0.f);
    }
  }
  __syncthreads();
  // ---- phase 3: head layer 1 (16x160 @ 160->64 per sensor), K chunked ----
  _Float16* h1b = (_Float16*)scratch;  // 8s*4nt*512 per K chunk
  v8f a0 = {}, a1 = {}, a2 = {}, a3 = {};
  for (int kt = 0; kt < 5; ++kt) {
    __syncthreads();
    for (int idx = tid; idx < 8 * 4 * 512; idx += 256) {
      int e = idx & 15, ln = (idx >> 4) & 31, fp = idx >> 9;
      int s = fp >> 2, nt = fp & 3;
      int n = nt * 16 + (ln & 15);
      int k = kt * 32 + (ln >> 4) * 8 + (e < 8 ? e : e + 8);
      h1b[idx] = (_Float16)hw1[(s * 64 + n) * 160 + k];
    }
    __syncthreads();
    v16h av = ldfrag(cfA + (wv * 5 + kt) * 512, lane);
    a0 = wmma_f16(av, ldfrag(h1b + (wv * 4 + 0) * 512, lane), a0);
    a1 = wmma_f16(av, ldfrag(h1b + (wv * 4 + 1) * 512, lane), a1);
    a2 = wmma_f16(av, ldfrag(h1b + (wv * 4 + 2) * 512, lane), a2);
    a3 = wmma_f16(av, ldfrag(h1b + (wv * 4 + 3) * 512, lane), a3);
  }
  {  // relu+bias -> h1 A fragments (K = 64 for next layer)
#pragma unroll
    for (int nt = 0; nt < 4; ++nt) {
      v8f& a = (nt == 0) ? a0 : (nt == 1) ? a1 : (nt == 2) ? a2 : a3;
      int j = nt * 16 + (lane & 15);  // 0..63
      int kth = j >> 5, hh, ee;
      a_inv(j & 31, hh, ee);
#pragma unroll
      for (int r = 0; r < 8; ++r)
        h1A[(wv * 2 + kth) * 512 + (mh + r + hh * 16) * 16 + ee] =
            (_Float16)fmaxf(a[r] + hb1[wv * 64 + j], 0.f);
    }
  }
  __syncthreads();
  // ---- phase 4: head layer 2 (16x64 @ 64->32 per sensor) ----
  _Float16* h2b = (_Float16*)scratch;  // 8s*2kt*2nt*512 = 32 KB, load once
  for (int idx = tid; idx < 8 * 2 * 2 * 512; idx += 256) {
    int e = idx & 15, ln = (idx >> 4) & 31, fp = idx >> 9;
    int s = fp >> 2, kt = (fp >> 1) & 1, nt = fp & 1;
    int n = nt * 16 + (ln & 15);
    int k = kt * 32 + (ln >> 4) * 8 + (e < 8 ? e : e + 8);
    h2b[idx] = (_Float16)hw2[(s * 32 + n) * 64 + k];
  }
  __syncthreads();
  {
    v8f c0 = {}, c1 = {};
    v16h av0 = ldfrag(h1A + (wv * 2 + 0) * 512, lane);
    v16h av1 = ldfrag(h1A + (wv * 2 + 1) * 512, lane);
    c0 = wmma_f16(av0, ldfrag(h2b + ((wv * 2 + 0) * 2 + 0) * 512, lane), c0);
    c0 = wmma_f16(av1, ldfrag(h2b + ((wv * 2 + 1) * 2 + 0) * 512, lane), c0);
    c1 = wmma_f16(av0, ldfrag(h2b + ((wv * 2 + 0) * 2 + 1) * 512, lane), c1);
    c1 = wmma_f16(av1, ldfrag(h2b + ((wv * 2 + 1) * 2 + 1) * 512, lane), c1);
#pragma unroll
    for (int nt = 0; nt < 2; ++nt) {
      v8f& c = nt ? c1 : c0;
      int u = nt * 16 + (lane & 15);
#pragma unroll
      for (int r = 0; r < 8; ++r)
        h2p[((mh + r) * 8 + wv) * 32 + u] = fmaxf(c[r] + hb2[wv * 32 + u], 0.f);
    }
  }
  __syncthreads();
  if (tid < 128) {  // head layer 3 + sigmoid
    int m = tid >> 3, s = tid & 7;
    const float* w = hw3 + s * 32;
    float z = hb3[s];
    for (int k = 0; k < 32; ++k) z += h2p[(m * 8 + s) * 32 + k] * w[k];
    out_anom[(size_t)(b0 + m) * 8 + s] = sigf(z);
  }
}

extern "C" void kernel_launch(void* const* d_in, const int* in_sizes, int n_in,
                              void* d_out, int out_size, void* d_ws, size_t ws_size,
                              hipStream_t stream) {
  const float* x      = (const float*)d_in[0];
  const float* conv_w = (const float*)d_in[1];
  const float* conv_b = (const float*)d_in[2];
  const float* gamma  = (const float*)d_in[3];
  const float* beta   = (const float*)d_in[4];
  const float* aw1    = (const float*)d_in[5];
  const float* ab1    = (const float*)d_in[6];
  const float* aw2    = (const float*)d_in[7];
  const float* ab2    = (const float*)d_in[8];
  const float* wihf   = (const float*)d_in[9];
  const float* whhf   = (const float*)d_in[10];
  const float* bf     = (const float*)d_in[11];
  const float* wihr   = (const float*)d_in[12];
  const float* whhr   = (const float*)d_in[13];
  const float* br     = (const float*)d_in[14];
  const float* tw1    = (const float*)d_in[15];
  const float* tb1    = (const float*)d_in[16];
  const float* tw2    = (const float*)d_in[17];
  const float* tb2    = (const float*)d_in[18];
  const float* fw1    = (const float*)d_in[19];
  const float* fb1    = (const float*)d_in[20];
  const float* fw2    = (const float*)d_in[21];
  const float* fb2    = (const float*)d_in[22];
  const float* cw1    = (const float*)d_in[23];
  const float* cb1    = (const float*)d_in[24];
  const float* cw2    = (const float*)d_in[25];
  const float* cb2    = (const float*)d_in[26];
  const float* sw     = (const float*)d_in[27];
  const float* sb     = (const float*)d_in[28];
  const float* hw1    = (const float*)d_in[29];
  const float* hb1    = (const float*)d_in[30];
  const float* hw2    = (const float*)d_in[31];
  const float* hb2    = (const float*)d_in[32];
  const float* hw3    = (const float*)d_in[33];
  const float* hb3    = (const float*)d_in[34];

  char* ws = (char*)d_ws;
  float*    y_ws   = (float*)ws;                 // 512*72*16 f32 = 2359296 B
  _Float16* seq_ws = (_Float16*)(ws + 2359296);  // 32*16*3*512 f16 frag = 1572864 B
  _Float16* raw_ws = (_Float16*)(ws + 3932160);  // 512*1024 f16 = 1048576 B
  float*    ctx_ws = (float*)(ws + 4980736);     // 512*64 f32 = 131072 B

  float* out = (float*)d_out;
  float* out_fault = out;          // (512,1)
  float* out_anom  = out + 512;    // (512,8)
  float* out_attn  = out + 4608;   // (512,8)
  float* out_tw    = out + 8704;   // (512,16,1)

  conv_pool_kernel<<<dim3(16, 512), 288, 0, stream>>>(x, conv_w, conv_b, gamma, beta, y_ws);
  attn_combine_kernel<<<512, 256, 0, stream>>>(y_ws, aw1, ab1, aw2, ab2, out_attn, seq_ws, raw_ws);
  lstm_kernel<<<32, 256, 0, stream>>>(seq_ws, wihf, whhf, bf, wihr, whhr, br,
                                      tw1, tb1, tw2, tb2, fw1, fb1, fw2, fb2,
                                      ctx_ws, out_fault, out_tw);
  cross_heads_kernel<<<32, 256, 0, stream>>>(raw_ws, cw1, cb1, cw2, cb2, sw, sb,
                                             hw1, hb1, hw2, hb2, hw3, hb3,
                                             ctx_ws, out_anom);
}
